// MultiHeadAttention_51565377356100
// MI455X (gfx1250) — compile-verified, matrix operand order checked by reading, not run on hardware
//
#include <hip/hip_runtime.h>
#include <hip/hip_bf16.h>

// ---------------------------------------------------------------------------
// Types for CDNA5 WMMA (wave32): v_wmma_f32_16x16x32_bf16
// ---------------------------------------------------------------------------
typedef __bf16 bf16;
typedef bf16  v8bf  __attribute__((ext_vector_type(8)));    // 16 bytes
typedef bf16  v16bf __attribute__((ext_vector_type(16)));   // 32 bytes (A/B frag)
typedef float v8f   __attribute__((ext_vector_type(8)));    // C/D frag

union Frag { v16bf v; v8bf h[2]; };

__device__ __forceinline__ bf16 f2bf(float f) {
    union { float f; unsigned u; } x; x.f = f;
    unsigned r = x.u + 0x7FFFu + ((x.u >> 16) & 1u);        // round-nearest-even
    unsigned short hs = (unsigned short)(r >> 16);
    union { unsigned short s; bf16 b; } y; y.s = hs;
    return y.b;
}

__device__ __forceinline__ float wave16_max(float v) {
#pragma unroll
    for (int m = 1; m <= 8; m <<= 1) v = fmaxf(v, __shfl_xor(v, m, 32));
    return v;                       // xor masks 1..8 stay within 16-lane halves
}
__device__ __forceinline__ float wave16_sum(float v) {
#pragma unroll
    for (int m = 1; m <= 8; m <<= 1) v += __shfl_xor(v, m, 32);
    return v;
}

// ---------------------------------------------------------------------------
// gfx1250 async global->LDS copy (ASYNCcnt path). Per-lane 16-byte copy:
//   LDS[vdst_lane] = MEM[vaddr_lane .. +15]
// LDS byte address = low 32 bits of the generic pointer (ISA aperture rule).
// ---------------------------------------------------------------------------
__device__ __forceinline__ unsigned lds_addr_of(const void* p) {
    return (unsigned)(unsigned long long)p;
}

__device__ __forceinline__ void async_g2l_b128(unsigned lds_addr, const void* gaddr) {
    asm volatile("global_load_async_to_lds_b128 %0, %1, off"
                 :: "v"(lds_addr), "v"(gaddr)
                 : "memory");
}

__device__ __forceinline__ void async_wait0() {
#if __has_builtin(__builtin_amdgcn_s_wait_asynccnt)
    __builtin_amdgcn_s_wait_asynccnt(0);
#else
    asm volatile("s_wait_asynccnt 0x0" ::: "memory");
#endif
}

// ---------------------------------------------------------------------------
// Elementwise helpers
// ---------------------------------------------------------------------------
__global__ void cast_f32_bf16_kernel(const float* __restrict__ in,
                                     bf16* __restrict__ out, int n) {
    int i = blockIdx.x * blockDim.x + threadIdx.x;
    if (i < n) out[i] = f2bf(in[i]);
}

__global__ void zero_f32_kernel(float* __restrict__ p, int n) {
    int i = blockIdx.x * blockDim.x + threadIdx.x;
    if (i < n) p[i] = 0.0f;
}

// ---------------------------------------------------------------------------
// Generic bf16 GEMM:  C[M,N] = A[M,K] @ B[K,N] + bias   (64x64 tile / block)
// 128 threads = 4 waves; wave w owns rows [w*16, w*16+16), 4 WMMA col-blocks.
// A tile arrives via async global->LDS; B tile is VGPR-staged (transpose).
// ---------------------------------------------------------------------------
template <int OUT_BF16>
__global__ __launch_bounds__(128) void gemm_bf16_kernel(
    const bf16* __restrict__ A, const bf16* __restrict__ B,
    const float* __restrict__ bias, void* __restrict__ C,
    int M, int N, int K)
{
    __shared__ __align__(16) bf16 sA[64 * 32];   // [m][k]  row-major
    __shared__ __align__(16) bf16 sB[64 * 32];   // [n][k]  transposed

    const int tid  = threadIdx.x;
    const int wave = tid >> 5;
    const int lane = tid & 31;
    const int tile_m = blockIdx.y * 64;
    const int tile_n = blockIdx.x * 64;
    const int kofs   = (lane < 16) ? 0 : 8;      // A/B 16-bit lane K layout

    v8f acc[4] = {v8f{0}, v8f{0}, v8f{0}, v8f{0}};

    for (int k0 = 0; k0 < K; k0 += 32) {
        __syncthreads();
        // --- A tile 64x32 (row-major): async global->LDS, b128 per lane ---
        for (int c = tid; c < 256; c += 128) {
            int r = c >> 2, kk = (c & 3) * 8;
            async_g2l_b128(lds_addr_of(&sA[r * 32 + kk]),
                           &A[(size_t)(tile_m + r) * K + k0 + kk]);
        }
        // --- B tile 32x64, transposed into [n][k] (VGPR-staged) -----------
        for (int c = tid; c < 256; c += 128) {
            int kk = c >> 3, n8 = (c & 7) * 8;
            v8bf t = *(const v8bf*)&B[(size_t)(k0 + kk) * N + tile_n + n8];
#pragma unroll
            for (int i = 0; i < 8; ++i) sB[(n8 + i) * 32 + kk] = t[i];
        }
        async_wait0();
        __syncthreads();

        // A fragment (16 rows of this wave, K=32)
        const int arow = wave * 16 + (lane & 15);
        Frag af;
        af.h[0] = *(const v8bf*)&sA[arow * 32 + kofs];
        af.h[1] = *(const v8bf*)&sA[arow * 32 + kofs + 16];
#pragma unroll
        for (int cb = 0; cb < 4; ++cb) {
            const int bcol = cb * 16 + (lane & 15);
            Frag bfr;
            bfr.h[0] = *(const v8bf*)&sB[bcol * 32 + kofs];
            bfr.h[1] = *(const v8bf*)&sB[bcol * 32 + kofs + 16];
            acc[cb] = __builtin_amdgcn_wmma_f32_16x16x32_bf16(
                false, af.v, false, bfr.v, (short)0, acc[cb], false, false);
        }
    }

    // Epilogue: C lane layout -> M = r + (lane<16?0:8), N = lane&15
    const int row0 = tile_m + wave * 16 + ((lane < 16) ? 0 : 8);
#pragma unroll
    for (int cb = 0; cb < 4; ++cb) {
        const int col = tile_n + cb * 16 + (lane & 15);
        const float bv = bias ? bias[col] : 0.0f;
#pragma unroll
        for (int r = 0; r < 8; ++r) {
            float v = acc[cb][r] + bv;
            if (OUT_BF16)
                ((bf16*)C)[(size_t)(row0 + r) * N + col] = f2bf(v);
            else
                ((float*)C)[(size_t)(row0 + r) * N + col] = v;
        }
    }
}

// ---------------------------------------------------------------------------
// Flash-style attention with exact per-head-summed attention output.
// Block = (batch b, head h, 64-row query tile). 128 threads = 4 waves,
// wave w owns query rows [w*16, w*16+16).
// Pass 1: streaming row max / exp-sum over all 16 K-tiles.
// Pass 2: normalized P via WMMA scores, atomic-add into attn_sum, P@V WMMA.
// K tile + mask tile arrive via async global->LDS; V tile transposed in VGPRs.
// ---------------------------------------------------------------------------
__global__ __launch_bounds__(128) void attn_kernel(
    const bf16* __restrict__ Qp, const bf16* __restrict__ Kp,
    const bf16* __restrict__ Vp, const unsigned char* __restrict__ mask,
    bf16* __restrict__ ctx, float* __restrict__ attn_sum, float scale)
{
    const int S = 1024, DP = 4096, SZ = 64, NT = S / 64;

    __shared__ __align__(16) bf16 sK[64 * 64];            // [krow][d]
    __shared__ __align__(16) bf16 sV[64 * 64];            // [d][krow] (transposed)
    __shared__ __align__(16) bf16 sP[64 * 64];            // [qrow][krow]
    __shared__ __align__(16) unsigned char sM[64 * 64];   // mask tile

    const int qt = blockIdx.x, h = blockIdx.y, b = blockIdx.z;
    const int tid = threadIdx.x, wave = tid >> 5, lane = tid & 31;
    const int kofs  = (lane < 16) ? 0 : 8;
    const int mhalf = (lane < 16) ? 0 : 8;
    const int q0 = qt * 64;

    const size_t headoff = (size_t)h * SZ;
    const bf16* Qb = Qp + (size_t)b * S * DP + headoff;
    const bf16* Kb = Kp + (size_t)b * S * DP + headoff;
    const bf16* Vb = Vp + (size_t)b * S * DP + headoff;

    // Q fragments for this wave's 16 rows, pinned in VGPRs (2 K-steps of 32)
    const int qrow = q0 + wave * 16 + (lane & 15);
    Frag qf[2];
#pragma unroll
    for (int s = 0; s < 2; ++s) {
        qf[s].h[0] = *(const v8bf*)&Qb[(size_t)qrow * DP + s * 32 + kofs];
        qf[s].h[1] = *(const v8bf*)&Qb[(size_t)qrow * DP + s * 32 + kofs + 16];
    }

    float mrow[8], lrow[8];
#pragma unroll
    for (int r = 0; r < 8; ++r) { mrow[r] = -__builtin_inff(); lrow[r] = 0.0f; }

    // ------------------------- pass 1: softmax stats -----------------------
    for (int kt = 0; kt < NT; ++kt) {
        const int k0 = kt * 64;
        __syncthreads();
        for (int c = tid; c < 512; c += 128) {               // K tile (async)
            int r = c >> 3, d8 = (c & 7) * 8;
            async_g2l_b128(lds_addr_of(&sK[r * 64 + d8]),
                           &Kb[(size_t)(k0 + r) * DP + d8]);
        }
        {                                                    // mask tile (async)
            const unsigned char* mp = mask + ((size_t)b * S + q0) * S + k0;
            for (int c = tid; c < 256; c += 128) {
                int r = c >> 2, k16 = (c & 3) * 16;
                async_g2l_b128(lds_addr_of(&sM[r * 64 + k16]),
                               &mp[(size_t)r * S + k16]);
            }
        }
        if (kt + 1 < NT)                                     // stream-ahead
            __builtin_prefetch(&Kb[(size_t)(k0 + 64 + (tid >> 1)) * DP + (tid & 1) * 32], 0, 3);
        async_wait0();
        __syncthreads();

        v8f sc[4] = {v8f{0}, v8f{0}, v8f{0}, v8f{0}};
#pragma unroll
        for (int s = 0; s < 2; ++s)
#pragma unroll
            for (int cb = 0; cb < 4; ++cb) {
                const int bcol = cb * 16 + (lane & 15);
                Frag kf;
                kf.h[0] = *(const v8bf*)&sK[bcol * 64 + s * 32 + kofs];
                kf.h[1] = *(const v8bf*)&sK[bcol * 64 + s * 32 + kofs + 16];
                sc[cb] = __builtin_amdgcn_wmma_f32_16x16x32_bf16(
                    false, qf[s].v, false, kf.v, (short)0, sc[cb], false, false);
            }

        float tmax[8];
#pragma unroll
        for (int r = 0; r < 8; ++r) tmax[r] = -__builtin_inff();
#pragma unroll
        for (int cb = 0; cb < 4; ++cb) {
            const int ncol = cb * 16 + (lane & 15);
#pragma unroll
            for (int r = 0; r < 8; ++r) {
                float sv = sc[cb][r] * scale;
                if (sM[(wave * 16 + mhalf + r) * 64 + ncol]) sv = -__builtin_inff();
                sc[cb][r] = sv;
                tmax[r] = fmaxf(tmax[r], sv);
            }
        }
#pragma unroll
        for (int r = 0; r < 8; ++r) {
            float nm = fmaxf(mrow[r], wave16_max(tmax[r]));
            float ps = 0.0f;
#pragma unroll
            for (int cb = 0; cb < 4; ++cb) {
                float sv = sc[cb][r];
                ps += (sv == -__builtin_inff()) ? 0.0f : __expf(sv - nm);
            }
            ps = wave16_sum(ps);
            float corr = (mrow[r] == -__builtin_inff()) ? 0.0f : __expf(mrow[r] - nm);
            lrow[r] = lrow[r] * corr + ps;
            mrow[r] = nm;
        }
    }

    // --------------------- pass 2: P, attn_sum, O = P@V --------------------
    v8f oacc[4] = {v8f{0}, v8f{0}, v8f{0}, v8f{0}};
    for (int kt = 0; kt < NT; ++kt) {
        const int k0 = kt * 64;
        __syncthreads();
        for (int c = tid; c < 512; c += 128) {               // K tile (async)
            int r = c >> 3, d8 = (c & 7) * 8;
            async_g2l_b128(lds_addr_of(&sK[r * 64 + d8]),
                           &Kb[(size_t)(k0 + r) * DP + d8]);
        }
        {                                                    // mask tile (async)
            const unsigned char* mp = mask + ((size_t)b * S + q0) * S + k0;
            for (int c = tid; c < 256; c += 128) {
                int r = c >> 2, k16 = (c & 3) * 16;
                async_g2l_b128(lds_addr_of(&sM[r * 64 + k16]),
                               &mp[(size_t)r * S + k16]);
            }
        }
        for (int c = tid; c < 512; c += 128) {               // V tile, transposed
            int r = c >> 3, d8 = (c & 7) * 8;
            v8bf t = *(const v8bf*)&Vb[(size_t)(k0 + r) * DP + d8];
#pragma unroll
            for (int i = 0; i < 8; ++i) sV[(d8 + i) * 64 + r] = t[i];
        }
        if (kt + 1 < NT)                                     // stream-ahead
            __builtin_prefetch(&Vb[(size_t)(k0 + 64 + (tid >> 1)) * DP + (tid & 1) * 32], 0, 3);
        async_wait0();
        __syncthreads();

        v8f sc[4] = {v8f{0}, v8f{0}, v8f{0}, v8f{0}};
#pragma unroll
        for (int s = 0; s < 2; ++s)
#pragma unroll
            for (int cb = 0; cb < 4; ++cb) {
                const int bcol = cb * 16 + (lane & 15);
                Frag kf;
                kf.h[0] = *(const v8bf*)&sK[bcol * 64 + s * 32 + kofs];
                kf.h[1] = *(const v8bf*)&sK[bcol * 64 + s * 32 + kofs + 16];
                sc[cb] = __builtin_amdgcn_wmma_f32_16x16x32_bf16(
                    false, qf[s].v, false, kf.v, (short)0, sc[cb], false, false);
            }

#pragma unroll
        for (int cb = 0; cb < 4; ++cb) {
            const int ncol = cb * 16 + (lane & 15);
#pragma unroll
            for (int r = 0; r < 8; ++r) {
                const int mr = wave * 16 + mhalf + r;
                float p = 0.0f;
                if (!sM[mr * 64 + ncol] && lrow[r] > 0.0f)
                    p = __expf(sc[cb][r] * scale - mrow[r]) / lrow[r];
                sP[mr * 64 + ncol] = f2bf(p);
                atomicAdd(&attn_sum[((size_t)b * S + q0 + mr) * S + k0 + ncol], p);
            }
        }
        __syncthreads();

#pragma unroll
        for (int s = 0; s < 2; ++s) {
            const int prow = wave * 16 + (lane & 15);
            Frag pf;
            pf.h[0] = *(const v8bf*)&sP[prow * 64 + s * 32 + kofs];
            pf.h[1] = *(const v8bf*)&sP[prow * 64 + s * 32 + kofs + 16];
#pragma unroll
            for (int cb = 0; cb < 4; ++cb) {
                const int dcol = cb * 16 + (lane & 15);
                Frag vf;
                vf.h[0] = *(const v8bf*)&sV[dcol * 64 + s * 32 + kofs];
                vf.h[1] = *(const v8bf*)&sV[dcol * 64 + s * 32 + kofs + 16];
                oacc[cb] = __builtin_amdgcn_wmma_f32_16x16x32_bf16(
                    false, pf.v, false, vf.v, (short)0, oacc[cb], false, false);
            }
        }
    }

    // ctx[b, q, h*64 + d] = O  (probabilities already normalized)
#pragma unroll
    for (int cb = 0; cb < 4; ++cb) {
        const int d = cb * 16 + (lane & 15);
#pragma unroll
        for (int r = 0; r < 8; ++r) {
            const int mr = q0 + wave * 16 + mhalf + r;
            ctx[((size_t)b * S + mr) * DP + headoff + d] = f2bf(oacc[cb][r]);
        }
    }
}

// ---------------------------------------------------------------------------
// Host-side orchestration
// ---------------------------------------------------------------------------
extern "C" void kernel_launch(void* const* d_in, const int* in_sizes, int n_in,
                              void* d_out, int out_size, void* d_ws, size_t ws_size,
                              hipStream_t stream) {
    (void)in_sizes; (void)n_in; (void)out_size; (void)ws_size;
    const int B = 2, S = 1024, IN = 512, DP = 4096;
    const int M = B * S;                       // 2048 rows

    const float* q  = (const float*)d_in[0];
    const float* k  = (const float*)d_in[1];
    const float* v  = (const float*)d_in[2];
    const unsigned char* mask = (const unsigned char*)d_in[3];
    const float* Wq = (const float*)d_in[4];
    const float* bq = (const float*)d_in[5];
    const float* Wk = (const float*)d_in[6];
    const float* bk = (const float*)d_in[7];
    const float* Wv = (const float*)d_in[8];
    const float* bv = (const float*)d_in[9];
    const float* Wf = (const float*)d_in[10];
    const float* bf = (const float*)d_in[11];

    float* out      = (float*)d_out;                       // [B,S,IN]
    float* attn_sum = out + (size_t)B * S * IN;            // [B,S,S]

    // Workspace layout (bf16 buffers, MB-aligned offsets)
    char* ws = (char*)d_ws;
    const size_t MB = 1ull << 20;
    bf16* qb  = (bf16*)(ws + 0 * MB);     // 2 MB each
    bf16* kb  = (bf16*)(ws + 2 * MB);
    bf16* vb  = (bf16*)(ws + 4 * MB);
    bf16* wqb = (bf16*)(ws + 6 * MB);     // 4 MB each
    bf16* wkb = (bf16*)(ws + 10 * MB);
    bf16* wvb = (bf16*)(ws + 14 * MB);
    bf16* wfb = (bf16*)(ws + 18 * MB);
    bf16* Qp  = (bf16*)(ws + 22 * MB);    // 16 MB each
    bf16* Kp  = (bf16*)(ws + 38 * MB);
    bf16* Vp  = (bf16*)(ws + 54 * MB);
    bf16* ctx = (bf16*)(ws + 70 * MB);

    const int nqkv = B * S * IN;          // 1,048,576
    const int nW   = IN * DP;             // 2,097,152
    cast_f32_bf16_kernel<<<nqkv / 256, 256, 0, stream>>>(q,  qb,  nqkv);
    cast_f32_bf16_kernel<<<nqkv / 256, 256, 0, stream>>>(k,  kb,  nqkv);
    cast_f32_bf16_kernel<<<nqkv / 256, 256, 0, stream>>>(v,  vb,  nqkv);
    cast_f32_bf16_kernel<<<nW / 256, 256, 0, stream>>>(Wq, wqb, nW);
    cast_f32_bf16_kernel<<<nW / 256, 256, 0, stream>>>(Wk, wkb, nW);
    cast_f32_bf16_kernel<<<nW / 256, 256, 0, stream>>>(Wv, wvb, nW);
    cast_f32_bf16_kernel<<<nW / 256, 256, 0, stream>>>(Wf, wfb, nW);

    // Projections: [2048,512] @ [512,4096] + bias -> bf16
    dim3 gp(DP / 64, M / 64);
    gemm_bf16_kernel<1><<<gp, 128, 0, stream>>>(qb, wqb, bq, Qp, M, DP, IN);
    gemm_bf16_kernel<1><<<gp, 128, 0, stream>>>(kb, wkb, bk, Kp, M, DP, IN);
    gemm_bf16_kernel<1><<<gp, 128, 0, stream>>>(vb, wvb, bv, Vp, M, DP, IN);

    // Zero the summed attention output (accumulated with atomics)
    const int natt = B * S * S;           // 2,097,152
    zero_f32_kernel<<<natt / 256, 256, 0, stream>>>(attn_sum, natt);

    // Attention: (16 q-tiles, 64 heads, 2 batches)
    dim3 ga(S / 64, 64, B);
    attn_kernel<<<ga, 128, 0, stream>>>(Qp, Kp, Vp, mask, ctx, attn_sum, 0.125f);

    // Final projection: [2048,4096] @ [4096,512] + bias -> fp32 out
    dim3 gf(IN / 64, M / 64);
    gemm_bf16_kernel<0><<<gf, 128, 0, stream>>>(ctx, wfb, bf, out, M, IN, DP);
}
